// OurRobertaSelfAttention_46849503265324
// MI455X (gfx1250) — matrix operand order repaired and checked by reading, not compiled, not run on hardware
//
#include <hip/hip_runtime.h>

// ---------------------------------------------------------------------------
// RoBERTa self-attention (relative_key) for gfx1250 / MI455X.
// bf16 WMMA (v_wmma_f32_16x16x32_bf16) everywhere, fp32 accumulate.
// Software-pipelined (double-buffered) fragment loads in all WMMA loops.
// B=8 S=512 H=1024 NH=16 HD=64 MAXP=512
// ---------------------------------------------------------------------------

typedef __attribute__((ext_vector_type(16))) __bf16        v16bf;
typedef __attribute__((ext_vector_type(8)))  float         v8f;
typedef __attribute__((ext_vector_type(4)))  unsigned int  u32x4;

union FragBF { v16bf v; u32x4 u[2]; };

#define WMMA_BF16(a, b, c) \
  __builtin_amdgcn_wmma_f32_16x16x32_bf16(false, (a), false, (b), (short)0, (c), false, false)

__device__ __forceinline__ unsigned short f32_to_bf16(float f) {
  union { float f; unsigned u; } x; x.f = f;
  unsigned r = x.u + 0x7FFFu + ((x.u >> 16) & 1u);   // round-to-nearest-even
  return (unsigned short)(r >> 16);
}

__device__ __forceinline__ v8f v8f_zero() {
  v8f r;
#pragma unroll
  for (int i = 0; i < 8; ++i) r[i] = 0.0f;
  return r;
}

// load a 16x32 bf16 A-fragment from a row-major [M,K] row pointer
__device__ __forceinline__ void load_a_frag(FragBF& f, const unsigned short* row,
                                            int k0, int akb) {
  f.u[0] = *(const u32x4*)(row + k0 + akb);
  f.u[1] = *(const u32x4*)(row + k0 + akb + 16);
}
// load a 32x16 bf16 B-fragment from a row-major [N,K] row pointer
__device__ __forceinline__ void load_b_frag(FragBF& f, const unsigned short* row,
                                            int k0, int bkb) {
  f.u[0] = *(const u32x4*)(row + k0 + bkb);
  f.u[1] = *(const u32x4*)(row + k0 + bkb + 16);
}

// ---------------------------------------------------------------------------
// fp32 -> bf16 convert / fill helpers
// ---------------------------------------------------------------------------
__global__ void cvt_f32_to_bf16(const float* __restrict__ src,
                                unsigned short* __restrict__ dst, int n) {
  int i = blockIdx.x * blockDim.x + threadIdx.x;
  if (i < n) dst[i] = f32_to_bf16(src[i]);
}

__global__ void fill_zero_u16(unsigned short* __restrict__ dst, int n) {
  int i = blockIdx.x * blockDim.x + threadIdx.x;
  if (i < n) dst[i] = 0;
}

// ---------------------------------------------------------------------------
// QKV projection: out[m,n] = sum_k X[m,k] * W[n,k] + bias[n]
// One wave computes a 16(M) x 128(N) tile; K = 1024 in 32 steps of 32.
// Double-buffered fragment loads so WMMAs overlap global loads.
// z = 0:Q, 1:K  -> [B,NH,S,HD] bf16 ;  z = 2:V -> transposed [B,NH,HD,S] bf16
// ---------------------------------------------------------------------------
#define NSUB 8   // 8 x 16 = 128 output columns per wave

__global__ __launch_bounds__(32) void qkv_proj_wmma(
    const unsigned short* __restrict__ X,     // [4096,1024] bf16
    const unsigned short* __restrict__ Wq,
    const unsigned short* __restrict__ Wk,
    const unsigned short* __restrict__ Wv,    // each [1024,1024] bf16
    const float* __restrict__ bq,
    const float* __restrict__ bk,
    const float* __restrict__ bv,
    unsigned short* __restrict__ Qo,          // [8,16,512,64]
    unsigned short* __restrict__ Ko,          // [8,16,512,64]
    unsigned short* __restrict__ Vt)          // [8,16,64,512]
{
  const int lane = threadIdx.x;
  const int mt   = blockIdx.x;                // 0..255 (token tiles)
  const int ng   = blockIdx.y;                // 0..7   (128-wide N groups)
  const int z    = blockIdx.z;                // 0..2

  const unsigned short* Wm  = (z == 0) ? Wq : (z == 1) ? Wk : Wv;
  const float*          bsp = (z == 0) ? bq : (z == 1) ? bk : bv;

  const int nl  = lane & 15;
  const int akb = (lane < 16) ? 0 : 8;        // A-frag K chunk base
  const int bkb = (lane < 16) ? 0 : 8;        // B-frag K chunk base
  const int mhi = (lane < 16) ? 0 : 8;        // C-frag M offset

  const unsigned short* xrow = X + (size_t)(mt * 16 + nl) * 1024;
  const unsigned short* wrow[NSUB];
#pragma unroll
  for (int s = 0; s < NSUB; ++s)
    wrow[s] = Wm + (size_t)(ng * (NSUB * 16) + s * 16 + nl) * 1024;

  v8f acc[NSUB];
#pragma unroll
  for (int s = 0; s < NSUB; ++s) acc[s] = v8f_zero();

  // prologue: load k-step 0
  FragBF a_cur, b_cur[NSUB];
  load_a_frag(a_cur, xrow, 0, akb);
#pragma unroll
  for (int s = 0; s < NSUB; ++s) load_b_frag(b_cur[s], wrow[s], 0, bkb);

  for (int kk = 0; kk < 32; ++kk) {
    const int kn = (kk + 1) & 31;             // wrap: last iter reloads k=0
    FragBF a_nxt, b_nxt[NSUB];
    load_a_frag(a_nxt, xrow, kn * 32, akb);   // issue next-step loads first
#pragma unroll
    for (int s = 0; s < NSUB; ++s) load_b_frag(b_nxt[s], wrow[s], kn * 32, bkb);

#pragma unroll
    for (int s = 0; s < NSUB; ++s)
      acc[s] = WMMA_BF16(a_cur.v, b_cur[s].v, acc[s]);

    a_cur = a_nxt;
#pragma unroll
    for (int s = 0; s < NSUB; ++s) b_cur[s] = b_nxt[s];
  }

#pragma unroll
  for (int s = 0; s < NSUB; ++s) {
    const int n     = ng * (NSUB * 16) + s * 16 + nl;
    const float bv_ = bsp[n];
    const int head  = n >> 6;
    const int d     = n & 63;
#pragma unroll
    for (int i = 0; i < 8; ++i) {
      const int m    = mt * 16 + mhi + i;
      const int bb   = m >> 9;                // batch
      const int sidx = m & 511;               // seq
      const unsigned short hv = f32_to_bf16(acc[s][i] + bv_);
      if (z == 0)
        Qo[((size_t)(bb * 16 + head) * 512 + sidx) * 64 + d] = hv;
      else if (z == 1)
        Ko[((size_t)(bb * 16 + head) * 512 + sidx) * 64 + d] = hv;
      else
        Vt[((size_t)(bb * 16 + head) * 64 + d) * 512 + sidx] = hv;
    }
  }
}

// ---------------------------------------------------------------------------
// Fused attention: one wave per (b, h, 16-query tile).
//   1) Wb = Q_tile @ E[l0 .. l0+527]^T   (rel-pos bias as WMMA, 16x528 in LDS)
//   2) scores = (Q K^T + gather(Wb)) / 8 + mask  -> Sc[16][512] (fp32 LDS)
//   3) row softmax -> Pb[16][512] bf16 (LDS)
//   4) ctx = P @ V   (V stored transposed [d, r])
// All global fragment loads are double-buffered across loop iterations.
// ---------------------------------------------------------------------------
#define WSTR 532   // fp32 stride of bias matrix  (bank-conflict padded)
#define SSTR 516   // fp32 stride of score matrix
#define PSTR 520   // bf16 stride of prob matrix

__global__ __launch_bounds__(32) void attn_wmma(
    const unsigned short* __restrict__ Q,    // [8,16,512,64]
    const unsigned short* __restrict__ K,    // [8,16,512,64]
    const unsigned short* __restrict__ VT,   // [8,16,64,512]
    const unsigned short* __restrict__ E,    // [1024,64] (row 1023 zero pad)
    const float* __restrict__ mask,          // [8,512]
    float* __restrict__ out)                 // [8,512,1024]
{
  __shared__ float          Wb[16 * WSTR];
  __shared__ float          Sc[16 * SSTR];
  __shared__ __align__(16) unsigned short Pb[16 * PSTR];

  const int lane = threadIdx.x;
  const int lt = blockIdx.x;                 // 0..31  query tile
  const int h  = blockIdx.y;                 // 0..15  head
  const int b  = blockIdx.z;                 // 0..7   batch
  const int l0 = lt * 16;

  const unsigned short* Qh = Q  + (size_t)(b * 16 + h) * 512 * 64;
  const unsigned short* Kh = K  + (size_t)(b * 16 + h) * 512 * 64;
  const unsigned short* Vh = VT + (size_t)(b * 16 + h) * 64 * 512;

  const int nl  = lane & 15;
  const int akb = (lane < 16) ? 0 : 8;
  const int bkb = (lane < 16) ? 0 : 8;
  const int mhi = (lane < 16) ? 0 : 8;

  // ---- Q tile as two A-fragments (K = d: 0..31, 32..63), kept in VGPRs ----
  FragBF qa0, qa1;
  {
    const unsigned short* qr = Qh + (size_t)(l0 + nl) * 64;
    load_a_frag(qa0, qr, 0, akb);
    load_a_frag(qa1, qr, 32, akb);
  }

  // ---- 1) relative-position bias: Wb[m][col] = q_m . E[l0 + col] ----------
  {
    const unsigned short* ebase = E + (size_t)(l0 + nl) * 64;
    FragBF e0c, e1c;
    load_b_frag(e0c, ebase, 0, bkb);
    load_b_frag(e1c, ebase, 32, bkb);
    for (int nt = 0; nt < 33; ++nt) {
      const int ntn = (nt + 1 < 33) ? nt + 1 : 0;
      const unsigned short* ern = ebase + (size_t)(ntn * 16) * 64;
      FragBF e0n, e1n;
      load_b_frag(e0n, ern, 0, bkb);
      load_b_frag(e1n, ern, 32, bkb);

      v8f c = v8f_zero();
      c = WMMA_BF16(qa0.v, e0c.v, c);
      c = WMMA_BF16(qa1.v, e1c.v, c);
#pragma unroll
      for (int i = 0; i < 8; ++i)
        Wb[(mhi + i) * WSTR + nt * 16 + nl] = c[i];

      e0c = e0n; e1c = e1n;
    }
  }
  __syncthreads();

  // ---- 2) scores = QK^T + bias gather, scaled + masked --------------------
  {
    const unsigned short* kbase = Kh + (size_t)nl * 64;
    FragBF k0c, k1c;
    load_b_frag(k0c, kbase, 0, bkb);
    load_b_frag(k1c, kbase, 32, bkb);
    for (int rt = 0; rt < 32; ++rt) {
      const int r0  = rt * 16;
      const int rtn = (rt + 1) & 31;
      const unsigned short* krn = kbase + (size_t)(rtn * 16) * 64;
      FragBF k0n, k1n;
      load_b_frag(k0n, krn, 0, bkb);
      load_b_frag(k1n, krn, 32, bkb);

      v8f c = v8f_zero();
      c = WMMA_BF16(qa0.v, k0c.v, c);
      c = WMMA_BF16(qa1.v, k1c.v, c);
      const float mv = mask[b * 512 + r0 + nl];
#pragma unroll
      for (int i = 0; i < 8; ++i) {
        const int m   = mhi + i;
        const int col = 511 + m - (r0 + nl);   // dist = l - r + 511, col 0..526
        Sc[m * SSTR + r0 + nl] = (c[i] + Wb[m * WSTR + col]) * 0.125f + mv;
      }
      k0c = k0n; k1c = k1n;
    }
  }
  __syncthreads();

  // ---- 3) softmax: 2 lanes per row, 256 columns each ----------------------
  {
    const int row = lane >> 1;
    const int off = (lane & 1) * 256;
    float* sr = Sc + row * SSTR + off;
    float mx = -3.402823466e38f;
    for (int t = 0; t < 256; ++t) mx = fmaxf(mx, sr[t]);
    mx = fmaxf(mx, __shfl_xor(mx, 1, 32));
    float sum = 0.0f;
    for (int t = 0; t < 256; ++t) {
      float e = __expf(sr[t] - mx);
      sr[t] = e;
      sum += e;
    }
    sum += __shfl_xor(sum, 1, 32);
    const float inv = 1.0f / sum;
    unsigned short* pr = Pb + row * PSTR + off;
    for (int t = 0; t < 256; ++t) pr[t] = f32_to_bf16(sr[t] * inv);
  }
  __syncthreads();

  // ---- 4) ctx = P @ V (A-frags from LDS, B-frags from V^T rows) -----------
  v8f acc[4];
#pragma unroll
  for (int s = 0; s < 4; ++s) acc[s] = v8f_zero();

  const unsigned short* vrow[4];
#pragma unroll
  for (int s = 0; s < 4; ++s) vrow[s] = Vh + (size_t)(s * 16 + nl) * 512;

  FragBF vb_cur[4];
#pragma unroll
  for (int s = 0; s < 4; ++s) load_b_frag(vb_cur[s], vrow[s], 0, bkb);

  for (int kc = 0; kc < 16; ++kc) {
    const int kn = (kc + 1) & 15;
    FragBF vb_nxt[4];
#pragma unroll
    for (int s = 0; s < 4; ++s) load_b_frag(vb_nxt[s], vrow[s], kn * 32, bkb);

    FragBF pa;
    const unsigned short* pr = Pb + nl * PSTR + kc * 32 + akb;
    pa.u[0] = *(const u32x4*)pr;
    pa.u[1] = *(const u32x4*)(pr + 16);
#pragma unroll
    for (int s = 0; s < 4; ++s)
      acc[s] = WMMA_BF16(pa.v, vb_cur[s].v, acc[s]);

#pragma unroll
    for (int s = 0; s < 4; ++s) vb_cur[s] = vb_nxt[s];
  }

#pragma unroll
  for (int s = 0; s < 4; ++s) {
#pragma unroll
    for (int i = 0; i < 8; ++i) {
      const int m = mhi + i;
      out[(size_t)(b * 512 + l0 + m) * 1024 + h * 64 + s * 16 + nl] = acc[s][i];
    }
  }
}

// ---------------------------------------------------------------------------
// Host launcher
// ---------------------------------------------------------------------------
extern "C" void kernel_launch(void* const* d_in, const int* in_sizes, int n_in,
                              void* d_out, int out_size, void* d_ws, size_t ws_size,
                              hipStream_t stream) {
  (void)in_sizes; (void)n_in; (void)out_size; (void)ws_size;

  const float* hidden = (const float*)d_in[0];   // [8,512,1024]
  const float* amask  = (const float*)d_in[1];   // [8,1,1,512]
  const float* Wq     = (const float*)d_in[2];
  const float* bq     = (const float*)d_in[3];
  const float* Wk     = (const float*)d_in[4];
  const float* bk     = (const float*)d_in[5];
  const float* Wv     = (const float*)d_in[6];
  const float* bv     = (const float*)d_in[7];
  const float* dist   = (const float*)d_in[8];   // [1023,64]
  float* out          = (float*)d_out;

  unsigned char* ws = (unsigned char*)d_ws;
  unsigned short* Xbf = (unsigned short*)(ws);                          // 8 MB
  unsigned short* Qbf = (unsigned short*)(ws + ( 8ull << 20));          // 8 MB
  unsigned short* Kbf = (unsigned short*)(ws + (16ull << 20));          // 8 MB
  unsigned short* Vtb = (unsigned short*)(ws + (24ull << 20));          // 8 MB
  unsigned short* Ebf = (unsigned short*)(ws + (32ull << 20));          // 128 KB (1024x64)
  unsigned short* Wqb = (unsigned short*)(ws + (32ull << 20) + (1u << 18));
  unsigned short* Wkb = Wqb + (1u << 20);
  unsigned short* Wvb = Wkb + (1u << 20);

  const int TC = 256;
  auto cvt = [&](const float* s, unsigned short* d, int n) {
    cvt_f32_to_bf16<<<dim3((n + TC - 1) / TC), dim3(TC), 0, stream>>>(s, d, n);
  };

  cvt(hidden, Xbf, 8 * 512 * 1024);
  cvt(Wq, Wqb, 1024 * 1024);
  cvt(Wk, Wkb, 1024 * 1024);
  cvt(Wv, Wvb, 1024 * 1024);
  cvt(dist, Ebf, 1023 * 64);
  fill_zero_u16<<<1, 64, 0, stream>>>(Ebf + 1023 * 64, 64);   // pad row 1023

  qkv_proj_wmma<<<dim3(256, 8, 3), dim3(32), 0, stream>>>(
      Xbf, Wqb, Wkb, Wvb, bq, bk, bv, Qbf, Kbf, Vtb);

  attn_wmma<<<dim3(32, 16, 8), dim3(32), 0, stream>>>(
      Qbf, Kbf, Vtb, Ebf, amask, out);
}
